// Context_ComplEx_v3_47399259078998
// MI455X (gfx1250) — matrix operand order repaired and checked by reading, not compiled, not run on hardware
//
#include <hip/hip_runtime.h>
#include <hip/hip_bf16.h>

// ---- CDNA5 WMMA vector types ------------------------------------------------
typedef __attribute__((ext_vector_type(16))) __bf16 v16bf;
typedef __attribute__((ext_vector_type(8)))  __bf16 bh8;
typedef __attribute__((ext_vector_type(8)))  float  v8f;

#define N_ENT_  400000
#define RANKK   128
#define TWOR    256
#define BATCH   512
#define NM      50

// GEMM kernel tiling
#define NT      8                // entity 16-tiles per block (processed in pairs)
#define PAIRS   (NT / 2)
#define AROW    264              // padded LDS row stride (bf16): 528B = 132 dw, rows -> banks 4r
#define ALDS    (BATCH * AROW)   // 135168 bf16 (qcat staged)
#define B2ROWS  32               // entity rows per pair-iteration
#define B2LDS   (B2ROWS * AROW)  // 8448 bf16 per pair buffer

__device__ __forceinline__ __bf16 f2bf(float f) {
    unsigned u = __builtin_bit_cast(unsigned, f);
    u = u + 0x7FFFu + ((u >> 16) & 1u);            // round-to-nearest-even
    unsigned short s = (unsigned short)(u >> 16);
    return __builtin_bit_cast(__bf16, s);
}

// ============================================================================
// Kernel 1: per-batch-element prep. One block (128 threads) per batch row.
// Produces f_lhs/f_rel/f_rhs/f_gec outputs and qcat=[qr|qi] in bf16 (workspace).
// ============================================================================
__global__ __launch_bounds__(128)
void batch_prep_kernel(const float* __restrict__ ent,
                       const float* __restrict__ rel,
                       const float* __restrict__ obj,
                       const float* __restrict__ W0,
                       const float* __restrict__ W1,
                       const float* __restrict__ bw0,
                       const float* __restrict__ bw1,
                       const float* __restrict__ Wo0,
                       const float* __restrict__ Uo0,
                       const float* __restrict__ Uo1,
                       const float* __restrict__ b_g,
                       const long long* __restrict__ x,
                       const long long* __restrict__ nb_idx,
                       float* __restrict__ f_lhs,
                       float* __restrict__ f_rel,
                       float* __restrict__ f_rhs,
                       float* __restrict__ f_gec,
                       __bf16* __restrict__ qws)
{
    __shared__ float s_lr[RANKK], s_li[RANKK], s_rr[RANKK], s_ri[RANKK];
    __shared__ float s_wr[RANKK], s_wi[RANKK];
    __shared__ float s_nb[NM * 257];     // padded stride 257 -> conflict free
    __shared__ float s_wnb[NM];
    __shared__ float s_red[RANKK];
    __shared__ float s_g;

    const int b = blockIdx.x;
    const int k = threadIdx.x;

    const long long si = x[b * 3 + 0];
    const long long pi = x[b * 3 + 1];
    const long long oi = x[b * 3 + 2];

    const float lr = ent[si * TWOR + k],        li = ent[si * TWOR + RANKK + k];
    const float rr = rel[pi * TWOR + k],        ri = rel[pi * TWOR + RANKK + k];
    const float hr = ent[oi * TWOR + k],        hi = ent[oi * TWOR + RANKK + k];
    s_lr[k] = lr; s_li[k] = li; s_rr[k] = rr; s_ri[k] = ri;

    f_lhs[b * RANKK + k] = sqrtf(lr * lr + li * li);
    f_rel[b * RANKK + k] = sqrtf(rr * rr + ri * ri);
    f_rhs[b * RANKK + k] = sqrtf(hr * hr + hi * hi);

    // gather neighbor embeddings into LDS
    for (int idx = k; idx < NM * TWOR; idx += RANKK) {
        const int m = idx / TWOR, c = idx - m * TWOR;
        s_nb[m * 257 + c] = obj[nb_idx[b * NM + m] * TWOR + c];
    }
    __syncthreads();

    // w_r, w_i : trp @ W0/W1 complex GEMM, column k
    float wr = bw0[k], wi = bw1[k];
#pragma unroll 4
    for (int j = 0; j < TWOR; ++j) {
        const float tr = (j < RANKK) ? s_lr[j] : s_rr[j - RANKK];
        const float ti = (j < RANKK) ? s_li[j] : s_ri[j - RANKK];
        const float w0 = W0[j * RANKK + k];
        const float w1 = W1[j * RANKK + k];
        wr += tr * w0 - ti * w1;
        wi += tr * w1 + ti * w0;
    }
    s_wr[k] = wr; s_wi[k] = wi;
    __syncthreads();

    // attention logits over neighbors
    if (k < NM) {
        float acc = 0.f;
        const float* nbm = &s_nb[k * 257];
        for (int j = 0; j < RANKK; ++j)
            acc += s_wr[j] * nbm[j] - s_wi[j] * nbm[RANKK + j];
        s_wnb[k] = acc;
    }
    __syncthreads();

    // softmax (serial over 50 by thread 0 — trivial)
    if (k == 0) {
        float mx = -INFINITY;
        for (int m = 0; m < NM; ++m) {
            float v = s_wnb[m];
            v = (v == 0.f) ? -INFINITY : v;
            s_wnb[m] = v;
            mx = fmaxf(mx, v);
        }
        float sum = 0.f;
        for (int m = 0; m < NM; ++m) { const float e = expf(s_wnb[m] - mx); s_wnb[m] = e; sum += e; }
        const float inv = 1.f / sum;
        for (int m = 0; m < NM; ++m) s_wnb[m] *= inv;
    }
    __syncthreads();

    // context vectors
    float ecr = 0.f, eci = 0.f;
#pragma unroll 5
    for (int m = 0; m < NM; ++m) {
        const float a = s_wnb[m];
        ecr += a * s_nb[m * 257 + k];
        eci += a * s_nb[m * 257 + RANKK + k];
    }

    const float srrr = lr * rr, siri = li * ri, sirr = li * rr, srri = lr * ri;
    const float c0 = srrr - siri, c1 = sirr + srri;

    // gate: block reduction of 128 partials
    s_red[k] = c0 * Uo0[k] - c1 * Uo1[k] + ecr * Wo0[k];
    __syncthreads();
    for (int s = RANKK / 2; s > 0; s >>= 1) {
        if (k < s) s_red[k] += s_red[k + s];
        __syncthreads();
    }
    if (k == 0) s_g = 1.f / (1.f + expf(-(s_red[0] + b_g[0])));
    __syncthreads();

    const float g  = s_g;
    const float gr = g * ecr + (1.f - g);
    const float gi = g * eci;
    const float qr = c0 * gr + c1 * gi;
    const float qi = c1 * gr - c0 * gi;

    f_gec[b * RANKK + k] = sqrtf(gr * gr + gi * gi);
    qws[b * TWOR + k]         = f2bf(qr);
    qws[b * TWOR + RANKK + k] = f2bf(qi);
}

// ============================================================================
// Kernel 2: scores = qcat(bf16) @ ent(bf16)^T via v_wmma_f32_16x16x32_bf16.
// Block = 256 threads = 8 waves. Block covers all 512 batch rows x 128 entities.
// Wave w: batch rows [64w, 64w+64) x 2 entity tiles per iteration
//   -> 8 accumulators, 12 ds_load_b128 feeding 8 WMMAs per K-chunk.
// ============================================================================
__global__ __launch_bounds__(256)
void scores_wmma_kernel(const float* __restrict__ ent,
                        const __bf16* __restrict__ qws,
                        float* __restrict__ scores)
{
    extern __shared__ char smem_raw[];
    __bf16* Abuf = (__bf16*)smem_raw;     // [512][AROW] qcat, padded
    __bf16* Bbuf = Abuf + ALDS;           // 2 x [32][AROW] ent tile pair, double buffered

    const int tid = threadIdx.x;
    const long long base = (long long)blockIdx.x * (NT * 16);

    // stage all of qcat (bf16, 256KB, L2-resident) into padded LDS rows
    for (int c = tid; c < BATCH * 32; c += 256) {
        const int row = c >> 5, seg = c & 31;
        *(uint4*)(Abuf + row * AROW + seg * 8) = *(const uint4*)(qws + row * TWOR + seg * 8);
    }
    // convert first entity tile-pair (32 rows) f32 -> bf16 into buffer 0
    {
        const int row = tid >> 3, col = (tid & 7) * 32;
        const float* src = ent + (base + row) * TWOR + col;
        union { __bf16 b[32]; uint4 u[4]; } tmp;
#pragma unroll
        for (int i = 0; i < 32; ++i) tmp.b[i] = f2bf(src[i]);
        __bf16* dst = Bbuf + row * AROW + col;
#pragma unroll
        for (int i = 0; i < 4; ++i) *(uint4*)(dst + i * 8) = tmp.u[i];
    }

    const int wave = tid >> 5, lane = tid & 31;
    const int g = lane >> 4, r16 = lane & 15;

    for (int it = 0; it < PAIRS; ++it) {
        __syncthreads();

        // prefetch+convert next tile-pair into the other buffer
        if (it + 1 < PAIRS) {
            const int row = tid >> 3, col = (tid & 7) * 32;
            const float* src = ent + (base + (it + 1) * B2ROWS + row) * TWOR + col;
            union { __bf16 b[32]; uint4 u[4]; } tmp;
#pragma unroll
            for (int i = 0; i < 32; ++i) tmp.b[i] = f2bf(src[i]);
            __bf16* dst = Bbuf + ((it + 1) & 1) * B2LDS + row * AROW + col;
#pragma unroll
            for (int i = 0; i < 4; ++i) *(uint4*)(dst + i * 8) = tmp.u[i];
        }

        const __bf16* Bb = Bbuf + (it & 1) * B2LDS + r16 * AROW;

        v8f acc[2][4] = {};   // [entity tile][batch tile] 16x16 f32 accumulators

#pragma unroll
        for (int chunk = 0; chunk < 8; ++chunk) {
            const int k0 = chunk * 32;
            v16bf Bop[2], Aop[4];

            // B operands (32x16, K x N): lane owns entity column r16;
            // half-wave 0 holds K=k0..k0+15, half-wave 1 holds K=k0+16..k0+31
#pragma unroll
            for (int u2 = 0; u2 < 2; ++u2) {
                const __bf16* p = Bb + u2 * (16 * AROW) + k0 + g * 16;
                const bh8 lo = *(const bh8*)(p);
                const bh8 hi = *(const bh8*)(p + 8);
                Bop[u2] = __builtin_shufflevector(lo, hi,
                    0, 1, 2, 3, 4, 5, 6, 7, 8, 9, 10, 11, 12, 13, 14, 15);
            }
            // A operands (16x32, M x K): lane owns batch row; K contiguous
            // to match B: half-wave0 {k0..k0+15}, half-wave1 {k0+16..k0+31}
#pragma unroll
            for (int t = 0; t < 4; ++t) {
                const __bf16* Ar = Abuf + (wave * 64 + t * 16 + r16) * AROW + k0;
                const bh8 lo = *(const bh8*)(Ar + g * 16);
                const bh8 hi = *(const bh8*)(Ar + g * 16 + 8);
                Aop[t] = __builtin_shufflevector(lo, hi,
                    0, 1, 2, 3, 4, 5, 6, 7, 8, 9, 10, 11, 12, 13, 14, 15);
            }
            // 8 back-to-back WMMAs reusing the batched operands
#pragma unroll
            for (int t = 0; t < 4; ++t) {
                acc[0][t] = __builtin_amdgcn_wmma_f32_16x16x32_bf16(
                    false, Aop[t], false, Bop[0], (short)0, acc[0][t], false, false);
                acc[1][t] = __builtin_amdgcn_wmma_f32_16x16x32_bf16(
                    false, Aop[t], false, Bop[1], (short)0, acc[1][t], false, false);
            }
        }

        // store 16x16 D tiles: lane holds column N=r16, rows e + 8*g
#pragma unroll
        for (int u2 = 0; u2 < 2; ++u2) {
            const long long ncol = base + it * B2ROWS + u2 * 16 + r16;
#pragma unroll
            for (int t = 0; t < 4; ++t) {
                const int mrow = wave * 64 + t * 16 + g * 8;
#pragma unroll
                for (int e = 0; e < 8; ++e) {
                    scores[(size_t)(mrow + e) * N_ENT_ + ncol] = acc[u2][t][e];
                }
            }
        }
    }
}

// ============================================================================
extern "C" void kernel_launch(void* const* d_in, const int* in_sizes, int n_in,
                              void* d_out, int out_size, void* d_ws, size_t ws_size,
                              hipStream_t stream)
{
    const float*     ent = (const float*)d_in[0];
    const float*     rel = (const float*)d_in[1];
    const float*     obj = (const float*)d_in[2];
    const float*     W0  = (const float*)d_in[3];
    const float*     W1  = (const float*)d_in[4];
    const float*     bw0 = (const float*)d_in[5];
    const float*     bw1 = (const float*)d_in[6];
    const float*     Wo0 = (const float*)d_in[7];
    const float*     Uo0 = (const float*)d_in[8];
    const float*     Uo1 = (const float*)d_in[9];
    const float*     b_g = (const float*)d_in[10];
    const long long* x   = (const long long*)d_in[11];
    const long long* nb  = (const long long*)d_in[12];

    float* out      = (float*)d_out;
    float* scores   = out;                                  // 512 x 400000
    float* f_lhs    = out + (size_t)BATCH * N_ENT_;
    float* f_rel_o  = f_lhs   + BATCH * RANKK;
    float* f_rhs_o  = f_rel_o + BATCH * RANKK;
    float* f_gec_o  = f_rhs_o + BATCH * RANKK;

    __bf16* qws = (__bf16*)d_ws;                            // 512 x 256 bf16

    batch_prep_kernel<<<BATCH, RANKK, 0, stream>>>(
        ent, rel, obj, W0, W1, bw0, bw1, Wo0, Uo0, Uo1, b_g, x, nb,
        f_lhs, f_rel_o, f_rhs_o, f_gec_o, qws);

    const size_t smem = (size_t)(ALDS + 2 * B2LDS) * sizeof(__bf16);  // 304128 B < 320KB/WGP
    scores_wmma_kernel<<<N_ENT_ / (NT * 16), 256, smem, stream>>>(ent, qws, scores);
}
